// SequenceCTM_81286551044521
// MI455X (gfx1250) — matrix operand order repaired and checked by reading, not compiled, not run on hardware
//
#include <hip/hip_runtime.h>
#include <hip/hip_bf16.h>

// ---------------- CDNA5 (gfx1250) types ----------------
typedef __bf16 bf16_t;
typedef __attribute__((ext_vector_type(16))) __bf16 v16bf;
typedef __attribute__((ext_vector_type(8)))  __bf16 v8bf;
typedef __attribute__((ext_vector_type(8)))  float  v8f;

#define BATCH   1024
#define SEQ     64
#define DMODEL  128
#define MEM     16
#define NTICKS  12
#define NHEADS  8
#define HDIM    16
#define NLMH    32
#define DSYNC   8256        // 128*129/2
#define NOUT    1000
#define NOUT_PAD 1024       // padded to multiple of 64 for the WMMA tiler

// ---------------- fragment loaders (ISA VGPR layouts) ----------------
// A (16x32 bf16): lanes 0-15: e0..7 -> K+0..7,  e8..15 -> K+16..23
//                 lanes16-31: e0..7 -> K+8..15, e8..15 -> K+24..31
__device__ __forceinline__ v16bf load_a_frag(const bf16_t* __restrict__ arow, int kk, int khalf)
{
    v8bf a0 = *(const v8bf*)(arow + kk + khalf * 8);
    v8bf a1 = *(const v8bf*)(arow + kk + 16 + khalf * 8);
    v16bf a;
#pragma unroll
    for (int e = 0; e < 8; ++e) { a[e] = a0[e]; a[8 + e] = a1[e]; }
    return a;
}
// B (32x16 bf16): lane holds column n = lane%16; same K order as A: e0..7 -> K+8*khalf+0..7, e8..15 -> K+16+8*khalf+0..7
__device__ __forceinline__ v16bf load_b_frag(const bf16_t* __restrict__ brow, int kk, int khalf)
{
    v8bf b0 = *(const v8bf*)(brow + kk + khalf * 8);
    v8bf b1 = *(const v8bf*)(brow + kk + 16 + khalf * 8);
    v16bf b;
#pragma unroll
    for (int e = 0; e < 8; ++e) { b[e] = b0[e]; b[8 + e] = b1[e]; }
    return b;
}

// =====================================================================
// bf16 WMMA GEMM:  C[M,N] = A[M,K] @ B[K,N] + bias[N]
// A row-major bf16 (M x K).  Bt = B^T row-major bf16 (Npad x K).
// One wave computes 32(M) x 64(N): 8 wmma / K-step.
// K-loop unrolled x2 with ping-pong register sets (no rotation copies):
//   load set1 @kk+32 ; wmma(set0) ; load set0 @kk+64 ; wmma(set1)
// Requires M % 32 == 0, K % 64 == 0, Bt padded to N multiple of 64.
// =====================================================================
template <int OUT_BF16>
__global__ __launch_bounds__(32)
void gemm_bf16_wmma(const bf16_t* __restrict__ A,
                    const bf16_t* __restrict__ Bt,
                    const float*  __restrict__ bias,
                    void* __restrict__ Cv,
                    int M, int N, int K,
                    long long strideM, long long strideN, long long cOff)
{
    const int lane  = threadIdx.x;          // 0..31
    const int l16   = lane & 15;
    const int khalf = lane >> 4;            // 0/1
    const int mTile = blockIdx.y * 32;
    const int nBase = blockIdx.x * 64;

    v8f acc[2][4];
#pragma unroll
    for (int mi = 0; mi < 2; ++mi)
#pragma unroll
        for (int j = 0; j < 4; ++j)
#pragma unroll
            for (int r = 0; r < 8; ++r) acc[mi][j][r] = 0.0f;

    const bf16_t* arow0 = A + (long long)(mTile + l16) * K;
    const bf16_t* arow1 = arow0 + (long long)16 * K;
    const bf16_t* brow[4];
#pragma unroll
    for (int j = 0; j < 4; ++j)
        brow[j] = Bt + (long long)(nBase + j * 16 + l16) * K;

    // ---- prologue: load K-step 0 into set0 ----
    v16bf a0_0 = load_a_frag(arow0, 0, khalf);
    v16bf a1_0 = load_a_frag(arow1, 0, khalf);
    v16bf b_0[4];
#pragma unroll
    for (int j = 0; j < 4; ++j) b_0[j] = load_b_frag(brow[j], 0, khalf);

    v16bf a0_1, a1_1, b_1[4];

    // ---- main loop: two K-steps (64) per iteration, peel the last pair ----
    const int kend = K - 64;
    int kk = 0;
    for (; kk < kend; kk += 64) {
        __builtin_prefetch(arow0 + kk + 1024, 0, 1); // global_prefetch_b8

        // load set1 @ kk+32
        a0_1 = load_a_frag(arow0, kk + 32, khalf);
        a1_1 = load_a_frag(arow1, kk + 32, khalf);
#pragma unroll
        for (int j = 0; j < 4; ++j) b_1[j] = load_b_frag(brow[j], kk + 32, khalf);

        // wmma on set0
#pragma unroll
        for (int j = 0; j < 4; ++j) {
            acc[0][j] = __builtin_amdgcn_wmma_f32_16x16x32_bf16(
                            false, a0_0, false, b_0[j], (short)0, acc[0][j], false, false);
            acc[1][j] = __builtin_amdgcn_wmma_f32_16x16x32_bf16(
                            false, a1_0, false, b_0[j], (short)0, acc[1][j], false, false);
        }

        // load set0 @ kk+64
        a0_0 = load_a_frag(arow0, kk + 64, khalf);
        a1_0 = load_a_frag(arow1, kk + 64, khalf);
#pragma unroll
        for (int j = 0; j < 4; ++j) b_0[j] = load_b_frag(brow[j], kk + 64, khalf);

        // wmma on set1
#pragma unroll
        for (int j = 0; j < 4; ++j) {
            acc[0][j] = __builtin_amdgcn_wmma_f32_16x16x32_bf16(
                            false, a0_1, false, b_1[j], (short)0, acc[0][j], false, false);
            acc[1][j] = __builtin_amdgcn_wmma_f32_16x16x32_bf16(
                            false, a1_1, false, b_1[j], (short)0, acc[1][j], false, false);
        }
    }
    // ---- peeled last 64: set0 holds @kk; load set1 @kk+32 ----
    a0_1 = load_a_frag(arow0, kk + 32, khalf);
    a1_1 = load_a_frag(arow1, kk + 32, khalf);
#pragma unroll
    for (int j = 0; j < 4; ++j) b_1[j] = load_b_frag(brow[j], kk + 32, khalf);
#pragma unroll
    for (int j = 0; j < 4; ++j) {
        acc[0][j] = __builtin_amdgcn_wmma_f32_16x16x32_bf16(
                        false, a0_0, false, b_0[j], (short)0, acc[0][j], false, false);
        acc[1][j] = __builtin_amdgcn_wmma_f32_16x16x32_bf16(
                        false, a1_0, false, b_0[j], (short)0, acc[1][j], false, false);
    }
#pragma unroll
    for (int j = 0; j < 4; ++j) {
        acc[0][j] = __builtin_amdgcn_wmma_f32_16x16x32_bf16(
                        false, a0_1, false, b_1[j], (short)0, acc[0][j], false, false);
        acc[1][j] = __builtin_amdgcn_wmma_f32_16x16x32_bf16(
                        false, a1_1, false, b_1[j], (short)0, acc[1][j], false, false);
    }

    // ---- store: C/D layout: lane -> col n=lane%16, VGPR r -> row r + 8*(lane/16)
#pragma unroll
    for (int j = 0; j < 4; ++j) {
        int n = nBase + j * 16 + l16;
        if (n >= N) continue;
        float bv = bias ? bias[n] : 0.0f;
#pragma unroll
        for (int mi = 0; mi < 2; ++mi) {
#pragma unroll
            for (int r = 0; r < 8; ++r) {
                int m = mTile + mi * 16 + r + khalf * 8;
                long long idx = cOff + (long long)m * strideM + (long long)n * strideN;
                float val = acc[mi][j][r] + bv;
                if (OUT_BF16) ((bf16_t*)Cv)[idx] = (bf16_t)val;
                else          ((float*)Cv)[idx]  = val;
            }
        }
    }
}

// =====================================================================
// Weight prep: W (K x N) f32 -> Wt (NpadRows x K) bf16, zero-padded rows
// =====================================================================
__global__ void conv_transpose_bf16(const float* __restrict__ W, bf16_t* __restrict__ Wt,
                                    int K, int N, int NpadRows)
{
    long long idx = (long long)blockIdx.x * blockDim.x + threadIdx.x;
    long long total = (long long)NpadRows * K;
    if (idx >= total) return;
    int k = (int)(idx % K);
    int n = (int)(idx / K);
    Wt[idx] = (n < N) ? (bf16_t)W[(long long)k * N + n] : (bf16_t)0.0f;
}

// =====================================================================
// Embedding
// =====================================================================
__global__ void embed_kernel(const int* __restrict__ numbers,
                             const float* __restrict__ num_embed,
                             const float* __restrict__ pos_embed,
                             bf16_t* __restrict__ embBf)
{
    int bs = blockIdx.x;
    int e  = threadIdx.x;
    int s  = bs & (SEQ - 1);
    int num = numbers[bs];
    float v = num_embed[(long long)num * DMODEL + e] + pos_embed[s * DMODEL + e];
    embBf[(long long)bs * DMODEL + e] = (bf16_t)v;
}

// =====================================================================
// State init: z, pre_hist broadcast; also seed bf16 z-half of concat buffer
// =====================================================================
__global__ void init_state(const float* __restrict__ z_init,
                           const float* __restrict__ pre_hist_init,
                           float* __restrict__ z, float* __restrict__ preHist,
                           bf16_t* __restrict__ hinBf)
{
    int b = blockIdx.x, d = threadIdx.x;
    float zv = z_init[d];
    z[b * DMODEL + d] = zv;
    hinBf[(long long)b * 256 + d] = (bf16_t)zv;
#pragma unroll
    for (int m = 0; m < MEM; ++m)
        preHist[((long long)b * DMODEL + d) * MEM + m] = pre_hist_init[d * MEM + m];
}

// =====================================================================
// Pair indices + decay tables
// =====================================================================
__global__ void pair_decay_prep(const float* __restrict__ decay,
                                unsigned char* __restrict__ idxI, unsigned char* __restrict__ idxJ,
                                float* __restrict__ ew, float* __restrict__ Dinv)
{
    int p = blockIdx.x * blockDim.x + threadIdx.x;
    if (p >= DSYNC) return;
    int i = 0, base = 0;
    while (p >= base + (DMODEL - i)) { base += DMODEL - i; ++i; }
    int j = i + (p - base);
    idxI[p] = (unsigned char)i;
    idxJ[p] = (unsigned char)j;
    float x = decay[p];
    float rate = (x > 20.f) ? x : log1pf(expf(x));
    float e = expf(-rate);
    ew[p] = e;
    float D = 0.f;
    for (int T = 1; T <= NTICKS + 1; ++T) {
        D = D * e + 1.0f;
        Dinv[T * DSYNC + p] = 1.0f / (D + 1e-8f);
    }
}

// N_1[b,p] = z_init[i]*z_init[j]
__global__ void init_N(const float* __restrict__ z_init,
                       const unsigned char* __restrict__ idxI, const unsigned char* __restrict__ idxJ,
                       float* __restrict__ Nbuf)
{
    int p = blockIdx.x * blockDim.x + threadIdx.x;
    int b = blockIdx.y;
    if (p >= DSYNC) return;
    Nbuf[(long long)b * DSYNC + p] = z_init[idxI[p]] * z_init[idxJ[p]];
}

// sync_bf[b,p] = bf16(N[b,p] * Dinv[T][p])  (used once for T=1)
__global__ void sync_scale(const float* __restrict__ Nbuf, const float* __restrict__ Dinv,
                           int T, bf16_t* __restrict__ syncBf)
{
    long long idx = (long long)blockIdx.x * blockDim.x + threadIdx.x;
    if (idx >= (long long)BATCH * DSYNC) return;
    int p = (int)(idx % DSYNC);
    syncBf[idx] = (bf16_t)(Nbuf[idx] * Dinv[T * DSYNC + p]);
}

// =====================================================================
// Attention: one block per (b,h), 64 threads; K/V in bf16
// =====================================================================
__global__ void attn_kernel(const float* __restrict__ Qf, const bf16_t* __restrict__ KV,
                            bf16_t* __restrict__ attnBf)
{
    int b = blockIdx.x, h = blockIdx.y;
    int s = threadIdx.x;
    __shared__ float sc[SEQ];
    __shared__ float red;
    const float*  q   = Qf + (long long)b * DMODEL + h * HDIM;
    const bf16_t* kvb = KV + ((long long)b * SEQ + s) * 256 + h * 32;
    float acc = 0.f;
#pragma unroll
    for (int d = 0; d < HDIM; ++d) acc += q[d] * (float)kvb[d];
    sc[s] = acc * 0.25f;               // 1/sqrt(16)
    __syncthreads();
    if (s == 0) { float m = sc[0]; for (int i = 1; i < SEQ; ++i) m = fmaxf(m, sc[i]); red = m; }
    __syncthreads();
    float e = expf(sc[s] - red);
    sc[s] = e;
    __syncthreads();
    if (s == 0) { float t = 0.f; for (int i = 0; i < SEQ; ++i) t += sc[i]; red = t; }
    __syncthreads();
    sc[s] = sc[s] / red;               // softmax weights
    __syncthreads();
    if (s < HDIM) {
        float o = 0.f;
        for (int i = 0; i < SEQ; ++i)
            o += sc[i] * (float)KV[((long long)b * SEQ + i) * 256 + h * 32 + HDIM + s];
        attnBf[(long long)b * DMODEL + h * HDIM + s] = (bf16_t)o;
    }
}

// LayerNorm(256) + ReLU -> bf16
__global__ void ln_relu(const float* __restrict__ h1,
                        const float* __restrict__ ln_g, const float* __restrict__ ln_b,
                        bf16_t* __restrict__ hBf)
{
    int b = blockIdx.x, i = threadIdx.x;
    __shared__ float s1[256], s2[256];
    float v = h1[(long long)b * 256 + i];
    s1[i] = v; s2[i] = v * v;
    __syncthreads();
    for (int off = 128; off > 0; off >>= 1) {
        if (i < off) { s1[i] += s1[i + off]; s2[i] += s2[i + off]; }
        __syncthreads();
    }
    float mu  = s1[0] * (1.0f / 256.0f);
    float var = s2[0] * (1.0f / 256.0f) - mu * mu;
    float nv  = (v - mu) * rsqrtf(var + 1e-5f) * ln_g[i] + ln_b[i];
    hBf[(long long)b * 256 + i] = (bf16_t)fmaxf(nv, 0.0f);
}

// =====================================================================
// Per-neuron NLM + history shift + z update (also writes bf16 z for next tick)
// =====================================================================
__global__ void nlm_update(const float* __restrict__ pre, float* __restrict__ preHist,
                           float* __restrict__ z, bf16_t* __restrict__ hinBf,
                           const float* __restrict__ nlm_w1, const float* __restrict__ nlm_b1,
                           const float* __restrict__ nlm_w2, const float* __restrict__ nlm_b2)
{
    int d  = blockIdx.x;
    int b  = blockIdx.y * 128 + threadIdx.x;
    int tb = threadIdx.x;
    __shared__ float w1s[MEM * NLMH];
    __shared__ float b1s[NLMH], w2s[NLMH];
    __shared__ float b2s;
    for (int i = tb; i < MEM * NLMH; i += 128) w1s[i] = nlm_w1[(long long)d * MEM * NLMH + i];
    if (tb < NLMH) { b1s[tb] = nlm_b1[d * NLMH + tb]; w2s[tb] = nlm_w2[d * NLMH + tb]; }
    if (tb == 0) b2s = nlm_b2[d];
    __syncthreads();

    float hist[MEM];
    float* hp = preHist + ((long long)b * DMODEL + d) * MEM;
#pragma unroll
    for (int m = 0; m < MEM - 1; ++m) hist[m] = hp[m + 1];
    hist[MEM - 1] = pre[(long long)b * DMODEL + d];
#pragma unroll
    for (int m = 0; m < MEM; ++m) hp[m] = hist[m];

    float zacc = b2s;
    for (int hh = 0; hh < NLMH; ++hh) {
        float a = b1s[hh];
#pragma unroll
        for (int m = 0; m < MEM; ++m) a += hist[m] * w1s[m * NLMH + hh];
        zacc += fmaxf(a, 0.0f) * w2s[hh];
    }
    z[(long long)b * DMODEL + d] = zacc;
    hinBf[(long long)b * 256 + d] = (bf16_t)zacc;   // z-half of concat for next tick
}

// Sync recurrence: N = ew*N + z_i*z_j ; fsync_bf = bf16(N * Dinv[T2])
__global__ void nsync_update(const float* __restrict__ z,
                             const unsigned char* __restrict__ idxI, const unsigned char* __restrict__ idxJ,
                             const float* __restrict__ ew, const float* __restrict__ Dinv, int T2,
                             float* __restrict__ Nbuf, bf16_t* __restrict__ fsyncBf)
{
    int b = blockIdx.y;
    int p = blockIdx.x * 256 + threadIdx.x;
    __shared__ float zs[DMODEL];
    if (threadIdx.x < DMODEL) zs[threadIdx.x] = z[(long long)b * DMODEL + threadIdx.x];
    __syncthreads();
    if (p >= DSYNC) return;
    long long idx = (long long)b * DSYNC + p;
    float n = ew[p] * Nbuf[idx] + zs[idxI[p]] * zs[idxJ[p]];
    Nbuf[idx] = n;
    fsyncBf[idx] = (bf16_t)(n * Dinv[T2 * DSYNC + p]);
}

// =====================================================================
// Host orchestration
// =====================================================================
extern "C" void kernel_launch(void* const* d_in, const int* in_sizes, int n_in,
                              void* d_out, int out_size, void* d_ws, size_t ws_size,
                              hipStream_t stream)
{
    (void)in_sizes; (void)n_in; (void)out_size; (void)ws_size;

    const int*   numbers       = (const int*)  d_in[0];
    const float* num_embed     = (const float*)d_in[1];
    const float* pos_embed     = (const float*)d_in[2];
    const float* kv_w          = (const float*)d_in[3];
    const float* kv_b          = (const float*)d_in[4];
    const float* z_init        = (const float*)d_in[5];
    const float* pre_hist_init = (const float*)d_in[6];
    const float* syn_w1        = (const float*)d_in[7];
    const float* syn_b1        = (const float*)d_in[8];
    const float* ln_g          = (const float*)d_in[9];
    const float* ln_b          = (const float*)d_in[10];
    const float* syn_w2        = (const float*)d_in[11];
    const float* syn_b2        = (const float*)d_in[12];
    const float* nlm_w1        = (const float*)d_in[13];
    const float* nlm_b1        = (const float*)d_in[14];
    const float* nlm_w2        = (const float*)d_in[15];
    const float* nlm_b2        = (const float*)d_in[16];
    const float* decay         = (const float*)d_in[17];
    const float* q_w           = (const float*)d_in[18];
    const float* q_b           = (const float*)d_in[19];
    const float* ao_w          = (const float*)d_in[20];
    const float* ao_b          = (const float*)d_in[21];
    const float* out_w         = (const float*)d_in[22];
    const float* out_b         = (const float*)d_in[23];
    float* out = (float*)d_out;

    // ---- workspace bump allocator (256B aligned) ----
    char* ws = (char*)d_ws;
    size_t off = 0;
    auto alloc = [&](size_t bytes) -> void* {
        void* p = ws + off;
        off += (bytes + 255) & ~(size_t)255;
        return p;
    };
    const long long BS = (long long)BATCH * SEQ;

    bf16_t* embBf  = (bf16_t*)alloc(BS * DMODEL * 2);
    bf16_t* KVbf   = (bf16_t*)alloc(BS * 256 * 2);
    bf16_t* kvwT   = (bf16_t*)alloc((size_t)256 * DMODEL * 2);
    bf16_t* qwT    = (bf16_t*)alloc((size_t)DMODEL * DSYNC * 2);
    bf16_t* aowT   = (bf16_t*)alloc((size_t)DMODEL * DMODEL * 2);
    bf16_t* sw1T   = (bf16_t*)alloc((size_t)256 * 256 * 2);
    bf16_t* sw2T   = (bf16_t*)alloc((size_t)DMODEL * 256 * 2);
    bf16_t* outwT  = (bf16_t*)alloc((size_t)NOUT_PAD * DSYNC * 2);
    float*  Nbuf   = (float*) alloc((size_t)BATCH * DSYNC * 4);
    bf16_t* fsyncBf= (bf16_t*)alloc((size_t)BATCH * DSYNC * 2);
    float*  Dinv   = (float*) alloc((size_t)(NTICKS + 2) * DSYNC * 4);
    float*  ewBuf  = (float*) alloc((size_t)DSYNC * 4);
    unsigned char* idxI = (unsigned char*)alloc(DSYNC);
    unsigned char* idxJ = (unsigned char*)alloc(DSYNC);
    float*  zBuf   = (float*) alloc((size_t)BATCH * DMODEL * 4);
    float*  preHist= (float*) alloc((size_t)BATCH * DMODEL * MEM * 4);
    float*  Qf     = (float*) alloc((size_t)BATCH * DMODEL * 4);
    bf16_t* attnBf = (bf16_t*)alloc((size_t)BATCH * DMODEL * 2);
    bf16_t* hinBf  = (bf16_t*)alloc((size_t)BATCH * 256 * 2);
    float*  h1     = (float*) alloc((size_t)BATCH * 256 * 4);
    bf16_t* hBf    = (bf16_t*)alloc((size_t)BATCH * 256 * 2);
    float*  preBuf = (float*) alloc((size_t)BATCH * DMODEL * 4);

    auto cvt = [&](const float* W, bf16_t* Wt, int K, int N, int rows) {
        long long tot = (long long)rows * K;
        conv_transpose_bf16<<<dim3((unsigned)((tot + 255) / 256)), 256, 0, stream>>>(W, Wt, K, N, rows);
    };

    // ---- one-time prep ----
    cvt(kv_w,   kvwT, DMODEL, 256,    256);
    cvt(q_w,    qwT,  DSYNC,  DMODEL, DMODEL);
    cvt(ao_w,   aowT, DMODEL, DMODEL, DMODEL);
    cvt(syn_w1, sw1T, 256,    256,    256);
    cvt(syn_w2, sw2T, 256,    DMODEL, DMODEL);
    cvt(out_w,  outwT, DSYNC, NOUT,   NOUT_PAD);

    pair_decay_prep<<<dim3((DSYNC + 255) / 256), 256, 0, stream>>>(decay, idxI, idxJ, ewBuf, Dinv);
    init_state<<<dim3(BATCH), DMODEL, 0, stream>>>(z_init, pre_hist_init, zBuf, preHist, hinBf);
    init_N<<<dim3((DSYNC + 255) / 256, BATCH), 256, 0, stream>>>(z_init, idxI, idxJ, Nbuf);

    embed_kernel<<<dim3((unsigned)BS), DMODEL, 0, stream>>>(numbers, num_embed, pos_embed, embBf);
    // KV projection: (B*S,128) @ (128,256) -> bf16
    gemm_bf16_wmma<1><<<dim3(256 / 64, (unsigned)(BS / 32)), 32, 0, stream>>>(
        embBf, kvwT, kv_b, KVbf, (int)BS, 256, DMODEL, 256, 1, 0);

    // sync for tick 0 (T=1); subsequent ticks reuse fsync (identical by recurrence)
    const long long syncTot = (long long)BATCH * DSYNC;
    sync_scale<<<dim3((unsigned)((syncTot + 255) / 256)), 256, 0, stream>>>(Nbuf, Dinv, 1, fsyncBf);

    for (int t = 0; t < NTICKS; ++t) {
        // Q = sync @ q_w + q_b : (1024,8256)@(8256,128) -> f32
        gemm_bf16_wmma<0><<<dim3(DMODEL / 64, BATCH / 32), 32, 0, stream>>>(
            fsyncBf, qwT, q_b, Qf, BATCH, DMODEL, DSYNC, DMODEL, 1, 0);
        // attention -> attnBf (bf16)
        attn_kernel<<<dim3(BATCH, NHEADS), SEQ, 0, stream>>>(Qf, KVbf, attnBf);
        // attn @ ao_w + ao_b -> bf16 directly into concat buffer's second half
        gemm_bf16_wmma<1><<<dim3(DMODEL / 64, BATCH / 32), 32, 0, stream>>>(
            attnBf, aowT, ao_b, hinBf, BATCH, DMODEL, DMODEL, 256, 1, 128);
        // syn1: (1024,256)@(256,256) -> f32
        gemm_bf16_wmma<0><<<dim3(256 / 64, BATCH / 32), 32, 0, stream>>>(
            hinBf, sw1T, syn_b1, h1, BATCH, 256, 256, 256, 1, 0);
        // LN + ReLU -> bf16
        ln_relu<<<dim3(BATCH), 256, 0, stream>>>(h1, ln_g, ln_b, hBf);
        // syn2: (1024,256)@(256,128) -> f32
        gemm_bf16_wmma<0><<<dim3(DMODEL / 64, BATCH / 32), 32, 0, stream>>>(
            hBf, sw2T, syn_b2, preBuf, BATCH, DMODEL, 256, DMODEL, 1, 0);
        // NLM + history shift + z update (+ bf16 z for next tick's concat)
        nlm_update<<<dim3(DMODEL, BATCH / 128), 128, 0, stream>>>(
            preBuf, preHist, zBuf, hinBf, nlm_w1, nlm_b1, nlm_w2, nlm_b2);
        // N recurrence + fsync (T = t+2) -> bf16 (also next tick's sync)
        nsync_update<<<dim3((DSYNC + 255) / 256, BATCH), 256, 0, stream>>>(
            zBuf, idxI, idxJ, ewBuf, Dinv, t + 2, Nbuf, fsyncBf);
        // out: (1024,8256)@(8256,1000) -> d_out[b,n,t]
        gemm_bf16_wmma<0><<<dim3(NOUT_PAD / 64, BATCH / 32), 32, 0, stream>>>(
            fsyncBf, outwT, out_b, out, BATCH, NOUT, DSYNC,
            (long long)NOUT * NTICKS, (long long)NTICKS, (long long)t);
    }
}
